// DTPViT_XL_16381005267314
// MI455X (gfx1250) — matrix operand order repaired and checked by reading, not compiled, not run on hardware
//
#include <hip/hip_runtime.h>

typedef __attribute__((ext_vector_type(16))) __bf16 v16bf;
typedef __attribute__((ext_vector_type(8)))  __bf16 v8bf;
typedef __attribute__((ext_vector_type(8)))  float  v8f;
typedef __attribute__((ext_vector_type(4))) unsigned int u32x4;
typedef __attribute__((ext_vector_type(8))) int i32x8;
typedef __attribute__((ext_vector_type(4))) int i32x4;

// ---------------- wave32 reductions ----------------
static __device__ __forceinline__ float wave_max(float v) {
#pragma unroll
  for (int o = 16; o; o >>= 1) v = fmaxf(v, __shfl_down(v, o, 32));
  return v;
}
static __device__ __forceinline__ float wave_sum(float v) {
#pragma unroll
  for (int o = 16; o; o >>= 1) v += __shfl_down(v, o, 32);
  return v;
}

// ---- A fragment (ISA 7.12.2): lane (M=lane&15) holds K {ab..ab+7} U {ab+16..ab+23}
static __device__ __forceinline__ v16bf load_afrag(const __bf16* p) {
  v8bf lo = *(const v8bf*)(p);
  v8bf hi = *(const v8bf*)(p + 16);
  return __builtin_shufflevector(lo, hi, 0, 1, 2, 3, 4, 5, 6, 7,
                                 8, 9, 10, 11, 12, 13, 14, 15);
}
// ---- B fragment from LDS row: K {kb..kb+7} U {kb+16..kb+23}, kb = 0|8
static __device__ __forceinline__ v16bf load_bfrag_lds(const __bf16* p) {
  v8bf lo = *(const v8bf*)(p);
  v8bf hi = *(const v8bf*)(p + 16);
  return __builtin_shufflevector(lo, hi, 0, 1, 2, 3, 4, 5, 6, 7,
                                 8, 9, 10, 11, 12, 13, 14, 15);
}

// ---- TDM: load 128 rows x 32 bf16 (row stride K elems) into LDS, 80B padded rows
// D# per ISA 8.3/8.4: count=1; data_size=2B; pad_enable, interval=16DW(64B), amount=4DW(16B)
static __device__ __forceinline__ void tdm_load_b_tile(const __bf16* gsrc,
                                                       unsigned lds_addr, int K) {
  unsigned long long ga = (unsigned long long)(size_t)gsrc;
  u32x4 g0;
  g0[0] = 1u;                                         // count=1 (valid descriptor)
  g0[1] = lds_addr;                                   // lds_addr
  g0[2] = (unsigned)ga;                               // global_addr[31:0]
  g0[3] = (unsigned)((ga >> 32) & 0x01FFFFFFu) | 0x80000000u;  // addr[56:32] | type=2
  i32x8 g1;
  g1[0] = 0x06D10000;          // data_size=1(2B) | pad_enable | interval=3(16DW) | amount=3(4DW)
  g1[1] = (int)(((unsigned)K & 0xFFFFu) << 16);       // tensor_dim0[15:0]
  g1[2] = (int)(((unsigned)K >> 16) & 0xFFFFu);       // tensor_dim0[31:16]; dim1 lo16=0
  g1[3] = (int)((1u << 4) | (32u << 16));             // dim1 hi16 (dim1=1<<20); tile_dim0=32
  g1[4] = 128;                                        // tile_dim1=128; tile_dim2=0
  g1[5] = K;                                          // tensor_dim0_stride[31:0]
  g1[6] = 0;
  g1[7] = 0;
  i32x4 gz = {0, 0, 0, 0};
  i32x8 gz8 = {0, 0, 0, 0, 0, 0, 0, 0};
  __builtin_amdgcn_tensor_load_to_lds(g0, g1, gz, gz, gz8, 0);
}

// ---------------- WMMA GEMM: C = A[M,K](bf16) @ W[N,K](bf16)^T + bias ----------------
// Block tile 128x128: 8 waves = 4M x 2N, wave tile 32x64 (8 wmma/k-step).
// B panel staged in LDS by the Tensor Data Mover, double buffered (TENSORcnt).
#define BROW 80   // padded LDS row stride (64B data + 16B pad)
__global__ __launch_bounds__(256) void gemm_wmma_tdm(
    const __bf16* __restrict__ A, const __bf16* __restrict__ W,
    const float* __restrict__ bias, float* __restrict__ C,
    __bf16* __restrict__ Cbf, int M, int N, int K, int gelu) {
  __shared__ __align__(16) char btile[2][128 * BROW];
  const int wave = threadIdx.x >> 5;
  const int lane = threadIdx.x & 31;
  const int mw = wave >> 1;            // 0..3
  const int nw = wave & 1;             // 0..1
  const int m0 = blockIdx.y * 128 + mw * 32;
  const int n0blk = blockIdx.x * 128;
  const int n0 = n0blk + nw * 64;
  const int l15 = lane & 15;

  int ar0 = m0 + l15;      if (ar0 >= M) ar0 = M - 1;
  int ar1 = m0 + 16 + l15; if (ar1 >= M) ar1 = M - 1;
  const int ab = (lane < 16) ? 0 : 8;
  const __bf16* ap0 = A + (size_t)ar0 * K + ab;
  const __bf16* ap1 = A + (size_t)ar1 * K + ab;

  const int bboff = (lane < 16) ? 0 : 16;   // byte offset of K base (0|8 elems) within LDS row
  const unsigned lds0 = (unsigned)(size_t)(&btile[0][0]);
  const unsigned lds1 = (unsigned)(size_t)(&btile[1][0]);
  const __bf16* wblk = W + (size_t)n0blk * K;

  if (wave == 0) tdm_load_b_tile(wblk, lds0, K);   // prologue: chunk 0

  v8f acc[2][4] = {};
  for (int kb = 0, it = 0; kb < K; kb += 32, ++it) {
    const int par = it & 1;
    __syncthreads();   // all waves done reading buffer (1-par) from previous iter
    if (wave == 0) {
      if (kb + 32 < K) {
        tdm_load_b_tile(wblk + kb + 32, par ? lds0 : lds1, K);
        __builtin_amdgcn_s_wait_tensorcnt(1);   // chunk 'it' complete
      } else {
        __builtin_amdgcn_s_wait_tensorcnt(0);
      }
    }
    __syncthreads();   // buffer 'par' ready
    const char* bbase = &btile[par][0];

    __builtin_prefetch(ap0 + kb + 64, 0, 1);
    const v16bf a0 = load_afrag(ap0 + kb);
    const v16bf a1 = load_afrag(ap1 + kb);
#pragma unroll
    for (int ni = 0; ni < 4; ++ni) {
      const int lcol = nw * 64 + ni * 16 + l15;
      const v16bf bq = load_bfrag_lds((const __bf16*)(bbase + lcol * BROW + bboff));
      acc[0][ni] = __builtin_amdgcn_wmma_f32_16x16x32_bf16(false, a0, false, bq, (short)0, acc[0][ni], false, false);
      acc[1][ni] = __builtin_amdgcn_wmma_f32_16x16x32_bf16(false, a1, false, bq, (short)0, acc[1][ni], false, false);
    }
  }

  // C/D layout: VGPR r -> M = tile_m + r + (lane<16 ? 0 : 8), N = tile_n + (lane&15)
  const int rb = (lane < 16) ? 0 : 8;
#pragma unroll
  for (int ni = 0; ni < 4; ++ni) {
    const int ccol = n0 + ni * 16 + l15;
    if (ccol >= N) continue;
    const float bv = bias ? bias[ccol] : 0.0f;
#pragma unroll
    for (int mi = 0; mi < 2; ++mi) {
#pragma unroll
      for (int r = 0; r < 8; ++r) {
        const int crow = m0 + mi * 16 + rb + r;
        if (crow < M) {
          float v = acc[mi][ni][r] + bv;
          if (gelu) v = 0.5f * v * (1.0f + erff(v * 0.70710678118654752f));
          if (C)   C[(size_t)crow * N + ccol] = v;
          if (Cbf) Cbf[(size_t)crow * N + ccol] = (__bf16)v;
        }
      }
    }
  }
}

// ---------------- elementwise f32 -> bf16 ----------------
__global__ void cvt_kernel(const float* __restrict__ in, __bf16* __restrict__ out, size_t n) {
  size_t i = (size_t)blockIdx.x * 256 + threadIdx.x;
  if (i < n) out[i] = (__bf16)in[i];
}

// ---------------- im2col for 16x16/16 patch conv -> [T*B, 768] bf16 ----------------
__global__ void im2col_kernel(const float* __restrict__ x, __bf16* __restrict__ out, int B) {
  const size_t total = (size_t)196 * B * 768;
  size_t idx = (size_t)blockIdx.x * 256 + threadIdx.x;
  if (idx >= total) return;
  int c = (int)(idx % 768);
  int row = (int)(idx / 768);
  int b = row % B, t = row / B;
  int py = t / 14, px = t % 14;
  int ch = c / 256, rem = c % 256, ky = rem / 16, kx = rem % 16;
  float v = x[(((size_t)b * 3 + ch) * 224 + (py * 16 + ky)) * 224 + (px * 16 + kx)];
  out[idx] = (__bf16)v;
}

// ---------------- sinusoidal relative position embedding (descending pos) ----------
__global__ void posemb_kernel(float* __restrict__ r, int T) {
  size_t idx = (size_t)blockIdx.x * 256 + threadIdx.x;
  if (idx >= (size_t)T * 768) return;
  int t = (int)(idx / 768), d = (int)(idx % 768);
  float pos = (float)(T - 1 - t);
  int k = (d < 384) ? d : d - 384;
  float inv = powf(10000.0f, -(2.0f * (float)k) / 768.0f);
  float s = pos * inv;
  r[idx] = (d < 384) ? sinf(s) : cosf(s);
}

// ---------------- fused rel-attention: scores + softmax + AV (bf16 out) ------------
__global__ __launch_bounds__(256) void attn_kernel(
    const float* __restrict__ qkv,   // [T*B, 2304] (q|k|v)
    const float* __restrict__ rk,    // [T, 768]
    const float* __restrict__ rwb, const float* __restrict__ rrb,  // [12,64]
    __bf16* __restrict__ out, int T, int B) {
  const int i = blockIdx.x, b = blockIdx.y, h = blockIdx.z;
  const int tid = threadIdx.x;
  __shared__ float qw[64], qr[64], qr2[64];
  __shared__ float sc[224];
  __shared__ float avs[4][64];
  __shared__ float red1[8], red2[8];

  if (tid < 64) {
    const float qv = qkv[((size_t)(i * B + b)) * 2304 + h * 64 + tid];
    qw[tid] = qv + rwb[h * 64 + tid];
    qr[tid] = qv + rrb[h * 64 + tid];
    float q2 = 0.0f;
    if (i + 1 < T) q2 = qkv[((size_t)((i + 1) * B + b)) * 2304 + h * 64 + tid];
    qr2[tid] = q2 + rrb[h * 64 + tid];
  }
  __syncthreads();

  float s = -3.0e38f;
  if (tid < T) {
    const int j = tid;
    const float* krow = qkv + ((size_t)(j * B + b)) * 2304 + 768 + h * 64;
    float ac = 0.0f;
#pragma unroll 8
    for (int d = 0; d < 64; ++d) ac += qw[d] * krow[d];
    float bd = 0.0f;
    if (j <= i) {                      // rel_shift: x[i, T-1+j-i]
      const float* rr = rk + (size_t)(T - 1 + j - i) * 768 + h * 64;
#pragma unroll 8
      for (int d = 0; d < 64; ++d) bd += qr[d] * rr[d];
    } else if (j > i + 1) {            // rel_shift wrap: x[i+1, j-i-2]
      const float* rr = rk + (size_t)(j - i - 2) * 768 + h * 64;
#pragma unroll 8
      for (int d = 0; d < 64; ++d) bd += qr2[d] * rr[d];
    }                                  // j == i+1 -> 0 (zero pad column)
    s = 0.125f * (ac + bd);
  }

  float m = wave_max(s);
  if ((tid & 31) == 0) red1[tid >> 5] = m;
  __syncthreads();
  if (tid == 0) { float mm = red1[0]; for (int w = 1; w < 8; ++w) mm = fmaxf(mm, red1[w]); red1[0] = mm; }
  __syncthreads();
  const float rowmax = red1[0];
  float e = (tid < T) ? expf(s - rowmax) : 0.0f;
  float ps = wave_sum(e);
  if ((tid & 31) == 0) red2[tid >> 5] = ps;
  __syncthreads();
  if (tid == 0) { float ss = 0.0f; for (int w = 0; w < 8; ++w) ss += red2[w]; red2[0] = ss; }
  __syncthreads();
  const float inv = 1.0f / red2[0];
  if (tid < T) sc[tid] = e * inv;
  __syncthreads();

  const int d = tid & 63, chunk = tid >> 6;
  float p = 0.0f;
  for (int j = chunk; j < T; j += 4)
    p += sc[j] * qkv[((size_t)(j * B + b)) * 2304 + 1536 + h * 64 + d];
  avs[chunk][d] = p;
  __syncthreads();
  if (tid < 64)
    out[((size_t)(i * B + b)) * 768 + h * 64 + tid] =
        (__bf16)(avs[0][tid] + avs[1][tid] + avs[2][tid] + avs[3][tid]);
}

// ---------------- (residual add +) LayerNorm, dual f32 + bf16 output ---------------
__global__ __launch_bounds__(256) void addln_kernel(
    const float* __restrict__ x, const float* __restrict__ resid,
    const float* __restrict__ g, const float* __restrict__ beta,
    float* __restrict__ y, __bf16* __restrict__ ybf) {
  const int row = blockIdx.x, tid = threadIdx.x;
  __shared__ float redS[8], redQ[8];
  const float* xr = x + (size_t)row * 768;
  const float* rr = resid ? resid + (size_t)row * 768 : nullptr;
  float v[3]; float s = 0.0f, q = 0.0f;
#pragma unroll
  for (int e = 0; e < 3; ++e) {
    const int d = tid + e * 256;
    float t = xr[d];
    if (rr) t += rr[d];
    v[e] = t; s += t; q += t * t;
  }
  s = wave_sum(s); q = wave_sum(q);
  if ((tid & 31) == 0) { redS[tid >> 5] = s; redQ[tid >> 5] = q; }
  __syncthreads();
  if (tid == 0) {
    float ts = 0, tq = 0;
    for (int w = 0; w < 8; ++w) { ts += redS[w]; tq += redQ[w]; }
    redS[0] = ts; redQ[0] = tq;
  }
  __syncthreads();
  const float mean = redS[0] * (1.0f / 768.0f);
  const float var = redQ[0] * (1.0f / 768.0f) - mean * mean;
  const float rstd = rsqrtf(var + 1e-5f);
#pragma unroll
  for (int e = 0; e < 3; ++e) {
    const int d = tid + e * 256;
    const float o = (v[e] - mean) * rstd * g[d] + beta[d];
    y[(size_t)row * 768 + d] = o;
    if (ybf) ybf[(size_t)row * 768 + d] = (__bf16)o;
  }
}

// ---------------- downsample: null token + segment means (interval 5) --------------
__global__ void pool_kernel(const float* __restrict__ h, const float* __restrict__ null_tok,
                            float* __restrict__ out, int B) {
  const int r = blockIdx.x, b = blockIdx.y, tid = threadIdx.x;
#pragma unroll
  for (int e = 0; e < 3; ++e) {
    const int d = tid + e * 256;
    float val;
    if (r == 0) {
      val = null_tok[d];
    } else {
      const int seg = r - 1;
      const int t0 = (seg == 0) ? 0 : 5 * seg - 4;
      const int t1 = 5 * seg;
      float sum = 0.0f;
      for (int t = t0; t <= t1; ++t) sum += h[((size_t)(t * B + b)) * 768 + d];
      val = sum / (float)(t1 - t0 + 1);
    }
    out[((size_t)(r * B + b)) * 768 + d] = val;
  }
}

// ---------------- mean over tokens -> [B, 768] ----------------
__global__ void meantok_kernel(const float* __restrict__ h, float* __restrict__ out,
                               int S, int B) {
  const int b = blockIdx.x, tid = threadIdx.x;
#pragma unroll
  for (int e = 0; e < 3; ++e) {
    const int d = tid + e * 256;
    float s = 0.0f;
    for (int t = 0; t < S; ++t) s += h[((size_t)(t * B + b)) * 768 + d];
    out[(size_t)b * 768 + d] = s / (float)S;
  }
}

// =====================================================================================
extern "C" void kernel_launch(void* const* d_in, const int* in_sizes, int n_in,
                              void* d_out, int out_size, void* d_ws, size_t ws_size,
                              hipStream_t stream) {
  (void)in_sizes; (void)n_in; (void)out_size; (void)ws_size;
  const float* x       = (const float*)d_in[0];
  const float* patch_w = (const float*)d_in[1];
  const float* patch_b = (const float*)d_in[2];
  const float* rwb     = (const float*)d_in[3];
  const float* rrb     = (const float*)d_in[4];
  const float* qkv_w   = (const float*)d_in[5];
  const float* qkv_b   = (const float*)d_in[6];
  const float* rnet_w  = (const float*)d_in[7];
  const float* rnet_b  = (const float*)d_in[8];
  const float* onet_w  = (const float*)d_in[9];
  const float* onet_b  = (const float*)d_in[10];
  const float* ln1_g   = (const float*)d_in[11];
  const float* ln1_b   = (const float*)d_in[12];
  const float* ff1_w   = (const float*)d_in[13];
  const float* ff1_b   = (const float*)d_in[14];
  const float* ff2_w   = (const float*)d_in[15];
  const float* ff2_b   = (const float*)d_in[16];
  const float* ln2_g   = (const float*)d_in[17];
  const float* ln2_b   = (const float*)d_in[18];
  const float* null_tok= (const float*)d_in[19];
  const float* down_g  = (const float*)d_in[20];
  const float* down_b  = (const float*)d_in[21];
  const float* head_w  = (const float*)d_in[22];
  const float* head_b  = (const float*)d_in[23];

  const int B = 32, T1 = 196, D = 768, FF = 3072, S1 = 41, L = 10;
  const int M1 = T1 * B;  // 6272
  const int M2 = S1 * B;  // 1312

  char* wsb = (char*)d_ws; size_t off = 0;
  auto alloc = [&](size_t bytes) -> void* {
    void* p = wsb + off; off = (off + bytes + 255) & ~(size_t)255; return p;
  };
  // activations
  float*  h     = (float*) alloc((size_t)M1 * D * 4);
  float*  bufC  = (float*) alloc((size_t)M1 * 3 * D * 4);
  __bf16* abuf  = (__bf16*)alloc((size_t)M1 * D * 2);
  __bf16* avbf  = (__bf16*)alloc((size_t)M1 * D * 2);
  __bf16* ffbuf = (__bf16*)alloc((size_t)M1 * FF * 2);
  float*  remb  = (float*) alloc((size_t)T1 * D * 4);
  __bf16* rbf   = (__bf16*)alloc((size_t)T1 * D * 2);
  float*  rkbuf = (float*) alloc((size_t)T1 * D * 4);
  float*  meanb = (float*) alloc((size_t)B * D * 4);
  // bf16 weights (converted once per launch; L2-resident thereafter)
  __bf16* pw    = (__bf16*)alloc((size_t)D * D * 2);
  __bf16* qkvw  = (__bf16*)alloc((size_t)L * 3 * D * D * 2);
  __bf16* rnetw = (__bf16*)alloc((size_t)L * D * D * 2);
  __bf16* onetw = (__bf16*)alloc((size_t)L * D * D * 2);
  __bf16* ff1w  = (__bf16*)alloc((size_t)L * FF * D * 2);
  __bf16* ff2w  = (__bf16*)alloc((size_t)L * D * FF * 2);
  __bf16* headw = (__bf16*)alloc((size_t)1000 * D * 2);
  (void)alloc((size_t)128 * FF * 2);   // TDM over-read pad after last weight buffer

  auto cvt = [&](const float* in, __bf16* outp, size_t n) {
    cvt_kernel<<<(unsigned)((n + 255) / 256), 256, 0, stream>>>(in, outp, n);
  };
  auto gemm = [&](const __bf16* A, const __bf16* W, const float* bias,
                  float* C, __bf16* Cbf, int M, int N, int K, int gelu) {
    dim3 grid((N + 127) / 128, (M + 127) / 128);
    gemm_wmma_tdm<<<grid, 256, 0, stream>>>(A, W, bias, C, Cbf, M, N, K, gelu);
  };

  // ---- convert all weights to bf16 once ----
  cvt(patch_w, pw,    (size_t)D * D);
  cvt(qkv_w,   qkvw,  (size_t)L * 3 * D * D);
  cvt(rnet_w,  rnetw, (size_t)L * D * D);
  cvt(onet_w,  onetw, (size_t)L * D * D);
  cvt(ff1_w,   ff1w,  (size_t)L * FF * D);
  cvt(ff2_w,   ff2w,  (size_t)L * D * FF);
  cvt(head_w,  headw, (size_t)1000 * D);

  // ---- patch embed: im2col + 768x768 GEMM -> h (f32) + abuf (bf16) ----
  {
    size_t total = (size_t)M1 * D;
    im2col_kernel<<<(unsigned)((total + 255) / 256), 256, 0, stream>>>(x, ffbuf, B);
    gemm(ffbuf, pw, patch_b, h, abuf, M1, D, D, 0);
  }

  auto run_layer = [&](int l, int M, int T) {
    gemm(abuf, qkvw + (size_t)l * 3 * D * D, qkv_b + (size_t)l * 3 * D,
         bufC, nullptr, M, 3 * D, D, 0);
    gemm(rbf, rnetw + (size_t)l * D * D, rnet_b + (size_t)l * D,
         rkbuf, nullptr, T, D, D, 0);
    attn_kernel<<<dim3(T, B, 12), 256, 0, stream>>>(bufC, rkbuf, rwb, rrb, avbf, T, B);
    gemm(avbf, onetw + (size_t)l * D * D, onet_b + (size_t)l * D,
         bufC, nullptr, M, D, D, 0);
    addln_kernel<<<M, 256, 0, stream>>>(bufC, h, ln1_g + (size_t)l * D,
                                        ln1_b + (size_t)l * D, h, abuf);
    gemm(abuf, ff1w + (size_t)l * FF * D, ff1_b + (size_t)l * FF,
         nullptr, ffbuf, M, FF, D, 1);                 // GELU fused, bf16 out
    gemm(ffbuf, ff2w + (size_t)l * D * FF, ff2_b + (size_t)l * D,
         bufC, nullptr, M, D, FF, 0);
    addln_kernel<<<M, 256, 0, stream>>>(bufC, h, ln2_g + (size_t)l * D,
                                        ln2_b + (size_t)l * D, h, abuf);
  };

  // ---- pre phase (T=196) ----
  {
    size_t n = (size_t)T1 * D;
    posemb_kernel<<<(unsigned)((n + 255) / 256), 256, 0, stream>>>(remb, T1);
    cvt(remb, rbf, n);
  }
  run_layer(0, M1, T1);
  run_layer(1, M1, T1);

  // ---- downsample: null token + segment means, then LN(down) ----
  pool_kernel<<<dim3(S1, B), 256, 0, stream>>>(h, null_tok, bufC, B);
  addln_kernel<<<M2, 256, 0, stream>>>(bufC, nullptr, down_g, down_b, h, abuf);

  // ---- short phase (T=41) ----
  {
    size_t n = (size_t)S1 * D;
    posemb_kernel<<<(unsigned)((n + 255) / 256), 256, 0, stream>>>(remb, S1);
    cvt(remb, rbf, n);
  }
  for (int l = 2; l < 10; ++l) run_layer(l, M2, S1);

  // ---- head: mean over tokens, then [32,768] @ [1000,768]^T ----
  meantok_kernel<<<B, 256, 0, stream>>>(h, meanb, S1, B);
  cvt(meanb, abuf, (size_t)B * D);
  gemm(abuf, headw, head_b, (float*)d_out, nullptr, B, 1000, D, 0);
}
